// RNN_8358006358147
// MI455X (gfx1250) — compile-verified, matrix operand order checked by reading, not run on hardware
//
#include <hip/hip_runtime.h>
#include <math.h>

typedef __attribute__((ext_vector_type(16))) _Float16 v16h;
typedef __attribute__((ext_vector_type(8)))  _Float16 v8h;
typedef __attribute__((ext_vector_type(8)))  float    v8f;

#define T_STEPS 2048
#define BATCH   64
#define DIN     128
#define HID     256
#define NCLS    5

#define HROW 264   // 256 + 8 halfs pad  (528 B row stride -> bank-conflict free)
#define WROW 264
#define XROW 136   // 128 + 8 halfs pad

// Async global->LDS (ASYNCcnt) path, gated on builtin availability (device pass only)
#if defined(__gfx1250__) && __has_builtin(__builtin_amdgcn_global_load_async_to_lds_b128)
#define HAS_ASYNC_LDS 1
#else
#define HAS_ASYNC_LDS 0
#endif

#if HAS_ASYNC_LDS
// Parameter types per hipcc diagnostic: (int4 AS1*, int4 AS3*, imm offset, imm cpol)
typedef int v4i_vs __attribute__((vector_size(16)));
typedef __attribute__((address_space(1))) v4i_vs* gasync_p;
typedef __attribute__((address_space(3))) v4i_vs* lasync_p;

__device__ __forceinline__ void async_wait_all() {
#if __has_builtin(__builtin_amdgcn_s_wait_asynccnt)
    __builtin_amdgcn_s_wait_asynccnt(0);
#else
    asm volatile("s_wait_asynccnt 0x0" ::: "memory");
#endif
}
// Each thread copies 64 B (32 halfs) of the 32 KB x_proj slice: 4 x b128 async ops.
__device__ __forceinline__ void async_copy_xp(const _Float16* gsrc,
                                              _Float16* lds_dst, int tid) {
    gasync_p g = (gasync_p)(gsrc + tid * 32);
    lasync_p l = (lasync_p)(lds_dst + tid * 32);
    __builtin_amdgcn_global_load_async_to_lds_b128(g, l, 0, 0);
    __builtin_amdgcn_global_load_async_to_lds_b128(g, l, 16, 0);
    __builtin_amdgcn_global_load_async_to_lds_b128(g, l, 32, 0);
    __builtin_amdgcn_global_load_async_to_lds_b128(g, l, 48, 0);
}
#endif

// ---------------------------------------------------------------------------
// Kernel 0: convert W_ih / W_hh f32 -> f16 into workspace
// ---------------------------------------------------------------------------
__global__ void cvt_weights_kernel(const float* __restrict__ Wih,
                                   const float* __restrict__ Whh,
                                   _Float16* __restrict__ Wih_h,
                                   _Float16* __restrict__ Whh_h) {
    int i = blockIdx.x * blockDim.x + threadIdx.x;
    if (i < HID * DIN) Wih_h[i] = (_Float16)Wih[i];
    if (i < HID * HID) Whh_h[i] = (_Float16)Whh[i];
}

// ---------------------------------------------------------------------------
// Kernel 1: x_proj[t,b,h] = inputs @ W_ih^T + b_ih + b_hh   (stored f16)
// 2048 blocks x 256 threads; each block: 64 rows x 256 cols, K=128 via WMMA.
// ---------------------------------------------------------------------------
__launch_bounds__(256)
__global__ void xproj_kernel(const float* __restrict__ inputs,
                             const _Float16* __restrict__ Wih_h,
                             const float* __restrict__ b_ih,
                             const float* __restrict__ b_hh,
                             _Float16* __restrict__ xproj) {
    __shared__ __align__(16) _Float16 sA[64 * XROW];   // 17.4 KB
    __shared__ __align__(16) _Float16 sW[HID * XROW];  // 69.6 KB

    const int tid  = threadIdx.x;
    const int lane = tid & 31;
    const int wave = tid >> 5;             // 0..7
    const size_t row0 = (size_t)blockIdx.x * 64;

    // Stage A tile: 64 rows x 128 f32 -> f16 (4 threads/row, float4 loads)
    {
        int r   = tid >> 2;
        int seg = (tid & 3) * 32;
        const float4* src = (const float4*)(inputs + (row0 + r) * DIN + seg);
        _Float16* dst = sA + r * XROW + seg;
        #pragma unroll
        for (int i = 0; i < 8; ++i) {
            float4 v = src[i];
            dst[4*i+0] = (_Float16)v.x; dst[4*i+1] = (_Float16)v.y;
            dst[4*i+2] = (_Float16)v.z; dst[4*i+3] = (_Float16)v.w;
        }
    }
    // Stage W_ih (already f16): one 128-half row per thread, b128 copies
    {
        const v8h* src = (const v8h*)(Wih_h + tid * DIN);
        v8h* dst = (v8h*)(sW + tid * XROW);
        #pragma unroll
        for (int i = 0; i < 16; ++i) dst[i] = src[i];
    }
    __syncthreads();

    const int tm       = wave & 3;          // 4 tile-rows (M)
    const int tnset    = wave >> 2;         // 0..1 -> 8 tile-cols each
    const int half_sel = lane >> 4;
    const int mlo      = lane & 15;

    v8f acc[8];
    #pragma unroll
    for (int i = 0; i < 8; ++i)
        #pragma unroll
        for (int j = 0; j < 8; ++j) acc[i][j] = 0.0f;

    const _Float16* arow = sA + (tm * 16 + mlo) * XROW;
    #pragma unroll
    for (int kk = 0; kk < 4; ++kk) {
        const int k0 = kk * 32 + half_sel * 8;
        union { v16h v; v8h h[2]; } ua;
        ua.h[0] = *(const v8h*)(arow + k0);
        ua.h[1] = *(const v8h*)(arow + k0 + 16);
        #pragma unroll
        for (int i = 0; i < 8; ++i) {
            const int tn = tnset * 8 + i;
            const _Float16* brow = sW + (tn * 16 + mlo) * XROW;
            union { v16h v; v8h h[2]; } ub;
            ub.h[0] = *(const v8h*)(brow + k0);
            ub.h[1] = *(const v8h*)(brow + k0 + 16);
            acc[i] = __builtin_amdgcn_wmma_f32_16x16x32_f16(
                false, ua.v, false, ub.v, (short)0, acc[i], false, false);
        }
    }

    #pragma unroll
    for (int i = 0; i < 8; ++i) {
        const int tn = tnset * 8 + i;
        const int n  = tn * 16 + mlo;
        const float bias = b_ih[n] + b_hh[n];
        #pragma unroll
        for (int r = 0; r < 8; ++r) {
            const size_t m = row0 + tm * 16 + half_sel * 8 + r;
            xproj[m * HID + n] = (_Float16)(acc[i][r] + bias);
        }
    }
}

// ---------------------------------------------------------------------------
// Kernel 2: sequential scan. ONE workgroup (16 wave32). W_hh + h double-buffer
// + x_proj double-buffer resident in LDS (~268 KB of the 320 KB WGP LDS).
// 512 WMMAs/step; next step's 32 KB x_proj slice streamed in with async
// global->LDS copies (ASYNCcnt) and absorbed under the WMMA work.
// ---------------------------------------------------------------------------
__launch_bounds__(512)
__global__ void rnn_scan_kernel(const _Float16* __restrict__ xproj,
                                const _Float16* __restrict__ Whh_h,
                                _Float16* __restrict__ h_out) {
    __shared__ __align__(16) _Float16 sW[HID * WROW];        // 135.2 KB
    __shared__ __align__(16) _Float16 sH[2][BATCH * HROW];   //  67.6 KB
#if HAS_ASYNC_LDS
    __shared__ __align__(16) _Float16 sXP[2][BATCH * HID];   //  64   KB
#endif

    const int tid  = threadIdx.x;
    const int lane = tid & 31;
    const int wave = tid >> 5;              // 0..15

    // Stage W_hh (f16) into padded LDS rows: 2 threads/row, 128 halfs each
    {
        const int row = tid >> 1;
        const int seg = (tid & 1) * 128;
        const v8h* src = (const v8h*)(Whh_h + row * HID + seg);
        v8h* dst = (v8h*)(sW + row * WROW + seg);
        #pragma unroll
        for (int i = 0; i < 16; ++i) dst[i] = src[i];
    }
    // h0 = 0 (both buffers)
    {
        _Float16* hp = &sH[0][0];
        for (int i = tid; i < 2 * BATCH * HROW; i += 512) hp[i] = (_Float16)0.0f;
    }
#if HAS_ASYNC_LDS
    async_copy_xp(xproj, &sXP[0][0], tid);   // stage t=0 slice
    async_wait_all();
#endif
    __syncthreads();

    const int tm       = wave >> 2;         // 0..3   (M tile row)
    const int tn0      = wave & 3;          // tiles tn0, tn0+4, tn0+8, tn0+12
    const int half_sel = lane >> 4;
    const int mlo      = lane & 15;

    int cur = 0;
    for (int t = 0; t < T_STEPS; ++t) {
        const _Float16* hbuf  = &sH[cur][0];
        _Float16*       hnext = &sH[cur ^ 1][0];

#if HAS_ASYNC_LDS
        if (t + 1 < T_STEPS)   // overlap next slice's copy with this step's WMMAs
            async_copy_xp(xproj + (size_t)(t + 1) * (BATCH * HID),
                          &sXP[(t + 1) & 1][0], tid);
        const _Float16* xsrc = &sXP[t & 1][0];
#else
        const _Float16* xsrc = xproj + (size_t)t * (BATCH * HID);
        if (t + 1 < T_STEPS)
            __builtin_prefetch((const void*)(xproj + (size_t)(t + 1) * (BATCH * HID) + tid * 32), 0, 1);
#endif

        v8f acc[4];
        #pragma unroll
        for (int i = 0; i < 4; ++i)
            #pragma unroll
            for (int j = 0; j < 8; ++j) acc[i][j] = 0.0f;

        const _Float16* arow = hbuf + (tm * 16 + mlo) * HROW;
        #pragma unroll
        for (int kk = 0; kk < 8; ++kk) {
            const int k0 = kk * 32 + half_sel * 8;
            union { v16h v; v8h h[2]; } ua;      // A fragment reused by 4 tiles
            ua.h[0] = *(const v8h*)(arow + k0);
            ua.h[1] = *(const v8h*)(arow + k0 + 16);
            #pragma unroll
            for (int i = 0; i < 4; ++i) {
                const int tn = tn0 + 4 * i;
                const _Float16* brow = sW + (tn * 16 + mlo) * WROW;
                union { v16h v; v8h h[2]; } ub;
                ub.h[0] = *(const v8h*)(brow + k0);
                ub.h[1] = *(const v8h*)(brow + k0 + 16);
                acc[i] = __builtin_amdgcn_wmma_f32_16x16x32_f16(
                    false, ua.v, false, ub.v, (short)0, acc[i], false, false);
            }
        }

        // h_new = tanh(acc + xp); store f16 into the other LDS buffer
        #pragma unroll
        for (int i = 0; i < 4; ++i) {
            const int n = (tn0 + 4 * i) * 16 + mlo;
            #pragma unroll
            for (int r = 0; r < 8; ++r) {
                const int m = tm * 16 + half_sel * 8 + r;
                const float s = acc[i][r] + (float)xsrc[m * HID + n];
                hnext[m * HROW + n] = (_Float16)tanhf(s);
            }
        }

#if HAS_ASYNC_LDS
        async_wait_all();      // next xp slice fully in LDS before the barrier
#endif
        __syncthreads();       // one barrier/step (double-buffered h and xp)
        cur ^= 1;
    }

    // export h_last (buffer `cur` holds the last write)
    {
        const _Float16* hf = &sH[cur][0];
        for (int i = tid; i < BATCH * HID; i += 512)
            h_out[i] = hf[(i >> 8) * HROW + (i & 255)];
    }
}

// ---------------------------------------------------------------------------
// Kernel 3: logits = h_last @ W_fc^T + b_fc ; log_softmax. One lane per batch.
// ---------------------------------------------------------------------------
__global__ void head_kernel(const _Float16* __restrict__ h_last,
                            const float* __restrict__ Wfc,
                            const float* __restrict__ bfc,
                            float* __restrict__ out) {
    int b = blockIdx.x * blockDim.x + threadIdx.x;
    if (b >= BATCH) return;
    float lg[NCLS];
    #pragma unroll
    for (int c = 0; c < NCLS; ++c) {
        float s = bfc[c];
        for (int k = 0; k < HID; ++k)
            s += (float)h_last[b * HID + k] * Wfc[c * HID + k];
        lg[c] = s;
    }
    float mx = lg[0];
    #pragma unroll
    for (int c = 1; c < NCLS; ++c) mx = fmaxf(mx, lg[c]);
    float se = 0.0f;
    #pragma unroll
    for (int c = 0; c < NCLS; ++c) se += expf(lg[c] - mx);
    const float lse = logf(se) + mx;
    #pragma unroll
    for (int c = 0; c < NCLS; ++c) out[b * NCLS + c] = lg[c] - lse;
}

// ---------------------------------------------------------------------------
extern "C" void kernel_launch(void* const* d_in, const int* in_sizes, int n_in,
                              void* d_out, int out_size, void* d_ws, size_t ws_size,
                              hipStream_t stream) {
    const float* inputs = (const float*)d_in[0];
    const float* W_ih   = (const float*)d_in[1];
    const float* W_hh   = (const float*)d_in[2];
    const float* b_ih   = (const float*)d_in[3];
    const float* b_hh   = (const float*)d_in[4];
    const float* W_fc   = (const float*)d_in[5];
    const float* b_fc   = (const float*)d_in[6];
    float* out = (float*)d_out;

    char* ws = (char*)d_ws;
    _Float16* Wih_h = (_Float16*)(ws);                                   //  64 KB
    _Float16* Whh_h = (_Float16*)(ws + 65536);                           // 128 KB
    _Float16* xproj = (_Float16*)(ws + 65536 + 131072);                  //  64 MB (fits L2)
    _Float16* hlast = (_Float16*)(ws + 65536 + 131072 + (size_t)67108864);

    cvt_weights_kernel<<<256, 256, 0, stream>>>(W_ih, W_hh, Wih_h, Whh_h);
    xproj_kernel<<<T_STEPS * BATCH / 64, 256, 0, stream>>>(inputs, Wih_h, b_ih, b_hh, xproj);
    rnn_scan_kernel<<<1, 512, 0, stream>>>(xproj, Whh_h, hlast);
    head_kernel<<<1, 64, 0, stream>>>(hlast, W_fc, b_fc, out);
}
